// ChunkedSlidingWindowAttention_1348619731137
// MI455X (gfx1250) — compile-verified, matrix operand order checked by reading, not run on hardware
//
#include <hip/hip_runtime.h>

// ---------------------------------------------------------------------------
// Types
// ---------------------------------------------------------------------------
typedef unsigned short ushort_t;
typedef __attribute__((ext_vector_type(16))) __bf16        bf16x16;
typedef __attribute__((ext_vector_type(8)))  float         f32x8;
typedef __attribute__((ext_vector_type(4)))  unsigned int  u32x4;
typedef __attribute__((ext_vector_type(8)))  int           i32x8;
typedef __attribute__((ext_vector_type(4)))  int           i32x4;

union BF16Frag {
    bf16x16 v;
    u32x4   q[2];
};

#define N_HEAD   16
#define HEAD_DIM 128
#define WINDOW   256
#define D_MODEL  2048
#define SEQ_T    8192

#if defined(__has_builtin)
#  if __has_builtin(__builtin_amdgcn_tensor_load_to_lds)
#    define HAVE_TDM 1
#  else
#    define HAVE_TDM 0
#  endif
#else
#  define HAVE_TDM 0
#endif

__device__ __forceinline__ void wait_tensorcnt0() {
#if defined(__has_builtin)
#  if __has_builtin(__builtin_amdgcn_s_wait_tensorcnt)
    __builtin_amdgcn_s_wait_tensorcnt(0);
    return;
#  endif
#endif
    asm volatile("s_wait_tensorcnt 0x0" ::: "memory");
}

__device__ __forceinline__ ushort_t f32_to_bf16(float f) {
    unsigned int u = __builtin_bit_cast(unsigned int, f);
    u += 0x7FFFu + ((u >> 16) & 1u);          // round-to-nearest-even
    return (ushort_t)(u >> 16);
}

// ---------------------------------------------------------------------------
// Kernel 1: fp32 -> bf16 conversion (grid-stride)
// ---------------------------------------------------------------------------
__global__ void cvt_f32_to_bf16(const float* __restrict__ src,
                                ushort_t* __restrict__ dst, int n) {
    int i = blockIdx.x * blockDim.x + threadIdx.x;
    int stride = gridDim.x * blockDim.x;
    for (; i < n; i += stride) dst[i] = f32_to_bf16(src[i]);
}

// ---------------------------------------------------------------------------
// Kernel 2: C[M,N] = A[M,K] * W[N,K]^T   (bf16 in, fp32 accum)
// wave tile 16x64, block = 4 waves -> 64x64 tiles.
// Register double-buffering with a peeled final step (no per-iteration
// zero-fill, no tail branch) + global_prefetch_b8 streaming.
// Optional fused RoPE epilogue (Q and K projections); fp32 or bf16 store.
// ---------------------------------------------------------------------------
__global__ void gemm_bf16_bt(const ushort_t* __restrict__ A,
                             const ushort_t* __restrict__ W,
                             ushort_t* __restrict__ outb,
                             float*    __restrict__ outf,
                             int K, int rope) {
    const int lane = threadIdx.x & 31;
    const int wave = threadIdx.x >> 5;
    const int half = lane >> 4;            // K-half select (wave32 A/B layout)
    const int l16  = lane & 15;
    const int rowBase = (blockIdx.y * 4 + wave) * 16;
    const int colBase = blockIdx.x * 64;

    const ushort_t* arow = A + (size_t)(rowBase + l16) * (size_t)K;
    const ushort_t* wr   = W + (size_t)(colBase + l16) * (size_t)K;

    const f32x8 zero = {0.f,0.f,0.f,0.f,0.f,0.f,0.f,0.f};
    f32x8 acc[4];
#pragma unroll
    for (int s = 0; s < 4; ++s) acc[s] = zero;

    // prologue: load fragments for k = 0
    BF16Frag a_c, b_c[4];
    {
        const int kA = half * 8;
        a_c.q[0] = *(const u32x4*)(arow + kA);
        a_c.q[1] = *(const u32x4*)(arow + kA + 16);
#pragma unroll
        for (int s = 0; s < 4; ++s) {
            const ushort_t* p = wr + (size_t)s * 16 * K + kA;
            b_c[s].q[0] = *(const u32x4*)(p);
            b_c[s].q[1] = *(const u32x4*)(p + 16);
        }
    }

    // steady state: always load k+32, WMMA on k (final step peeled below)
    for (int k0 = 0; k0 < K - 32; k0 += 32) {
        const int kA = k0 + 32 + half * 8;
        BF16Frag a_n, b_n[4];
        a_n.q[0] = *(const u32x4*)(arow + kA);
        a_n.q[1] = *(const u32x4*)(arow + kA + 16);
#pragma unroll
        for (int s = 0; s < 4; ++s) {
            const ushort_t* p = wr + (size_t)s * 16 * K + kA;
            b_n[s].q[0] = *(const u32x4*)(p);
            b_n[s].q[1] = *(const u32x4*)(p + 16);
        }
        // stream-ahead prefetch, 512B ahead of the in-flight loads
        if (k0 + 288 < K) {
            __builtin_prefetch(arow + k0 + 288, 0, 3);
#pragma unroll
            for (int s = 0; s < 4; ++s)
                __builtin_prefetch(wr + (size_t)s * 16 * K + k0 + 288, 0, 3);
        }
#pragma unroll
        for (int s = 0; s < 4; ++s) {
            acc[s] = __builtin_amdgcn_wmma_f32_16x16x32_bf16(
                false, a_c.v, false, b_c[s].v, (short)0, acc[s], false, false);
        }
        a_c = a_n;
#pragma unroll
        for (int s = 0; s < 4; ++s) b_c[s] = b_n[s];
    }
    // peeled final k-step
#pragma unroll
    for (int s = 0; s < 4; ++s) {
        acc[s] = __builtin_amdgcn_wmma_f32_16x16x32_bf16(
            false, a_c.v, false, b_c[s].v, (short)0, acc[s], false, false);
    }

    // Epilogue: D layout -> element (m = j + 8*half, n = l16)
#pragma unroll
    for (int s = 0; s < 4; ++s) {
#pragma unroll
        for (int j = 0; j < 8; ++j) {
            const int m   = j + half * 8;
            const int row = rowBase + m;
            const int col = colBase + s * 16 + l16;
            float v = acc[s][j];
            if (rope) {
                // pair partner col^1 lives in lane^1 (same fragment slot)
                float other = __shfl_xor(v, 1, 32);
                const int t = row & (SEQ_T - 1);
                const int d = col & (HEAD_DIM - 1);
                float ang = (float)t *
                            __powf(10000.0f, -(float)(d & ~1) * (1.0f / 128.0f));
                float sn, cs;
                __sincosf(ang, &sn, &cs);
                v = (d & 1) ? (other * sn + v * cs)   // xi' = xr*s + xi*c
                            : (v * cs - other * sn);  // xr' = xr*c - xi*s
            }
            const size_t idx = (size_t)row * D_MODEL + col;
            if (outf) outf[idx] = v;
            else      outb[idx] = f32_to_bf16(v);
        }
    }
}

// ---------------------------------------------------------------------------
// Kernel 3: chunked sliding-window flash attention.
// Block = (head, chunk, batch), 512 threads = 16 waves, one 16-query tile
// per wave. 512-key context streamed through LDS 32 keys at a time.
// K tile staged by the Tensor Data Mover (tensor_load_to_lds, TENSORcnt);
// V tile transposed manually into LDS so P*V B-fragments are contiguous.
// ---------------------------------------------------------------------------
__global__ __launch_bounds__(512)
void attn_swa(const ushort_t* __restrict__ Qb,
              const ushort_t* __restrict__ Kb,
              const ushort_t* __restrict__ Vb,
              ushort_t* __restrict__ Ob) {
    __shared__ __align__(16) ushort_t k_lds[32][128];    // [key][d]
    __shared__ __align__(16) ushort_t vT_lds[128][32];   // [d][key] (transposed)
    __shared__ __align__(16) ushort_t p_lds[16][16][32]; // per-wave P staging

    const int h    = blockIdx.x;
    const int c    = blockIdx.y;
    const int b    = blockIdx.z;
    const int tid  = threadIdx.x;
    const int lane = tid & 31;
    const int wave = tid >> 5;          // q-tile 0..15
    const int half = lane >> 4;
    const int l16  = lane & 15;

    // Resident Q fragments (16 rows x 128 d, rope already applied)
    const size_t qrow = ((size_t)b * SEQ_T + (size_t)c * WINDOW + wave * 16 + l16)
                        * D_MODEL + (size_t)h * HEAD_DIM;
    BF16Frag aq[4];
#pragma unroll
    for (int ds = 0; ds < 4; ++ds) {
        const ushort_t* p = Qb + qrow + ds * 32 + half * 8;
        aq[ds].q[0] = *(const u32x4*)(p);
        aq[ds].q[1] = *(const u32x4*)(p + 16);
    }

    const f32x8 zero = {0.f,0.f,0.f,0.f,0.f,0.f,0.f,0.f};
    f32x8 o[8];
#pragma unroll
    for (int nt = 0; nt < 8; ++nt) o[nt] = zero;
    float m_run[8], l_run[8];
#pragma unroll
    for (int j = 0; j < 8; ++j) { m_run[j] = -1e30f; l_run[j] = 0.f; }

    const float scale = 0.08838834764831845f;  // 1/sqrt(128)
    const int step0 = (c == 0) ? 8 : 0;        // chunk 0: prev-chunk keys masked

    for (int step = step0; step < 16; ++step) {
        __syncthreads();
        // --- first key row of this 32-key step (step is entirely in one chunk)
        const int jk0 = step * 32;
        const int t0  = (jk0 < WINDOW) ? ((c - 1) * WINDOW + jk0)
                                       : (c * WINDOW + jk0 - WINDOW);

#if HAVE_TDM
        // K tile: one TDM descriptor, issued by wave 0 (EXEC-independent DMA).
        // 2D tile: tile_dim0=128 (d, contiguous), tile_dim1=32 (keys),
        // tensor_dim0_stride = D_MODEL, data_size = 2B.
        if (wave == 0) {
            const unsigned lds_off = (unsigned)(uintptr_t)(&k_lds[0][0]);
            const unsigned long long ga = (unsigned long long)(uintptr_t)
                (Kb + ((size_t)b * SEQ_T + t0) * D_MODEL + (size_t)h * HEAD_DIM);
            u32x4 g0;
            g0[0] = 1u;                                   // count=1 (valid D#)
            g0[1] = lds_off;                              // lds_addr
            g0[2] = (unsigned)(ga & 0xFFFFFFFFu);         // global_addr[31:0]
            g0[3] = (unsigned)((ga >> 32) & 0x01FFFFFFu)  // global_addr[56:32]
                    | (2u << 30);                         // type = 2 ("image")
            i32x8 g1;
            g1[0] = (int)(1u << 16);        // data_size=1 -> 2 bytes
            g1[1] = (int)(128u << 16);      // tensor_dim0 = 128  (bits 79:48)
            g1[2] = (int)(32u  << 16);      // tensor_dim1 = 32   (bits 111:80)
            g1[3] = (int)(128u << 16);      // tile_dim0   = 128  (bits 127:112)
            g1[4] = 32;                     // tile_dim1   = 32   (bits 143:128)
            g1[5] = D_MODEL;                // tensor_dim0_stride (bits 207:160)
            g1[6] = 0;
            g1[7] = 0;
            const i32x4 gz = {0, 0, 0, 0};
#if __clang_major__ >= 23
            const i32x8 gz8 = {0, 0, 0, 0, 0, 0, 0, 0};
            __builtin_amdgcn_tensor_load_to_lds(g0, g1, gz, gz, gz8, 0);
#else
            __builtin_amdgcn_tensor_load_to_lds(g0, g1, gz, gz, 0);
#endif
            wait_tensorcnt0();
        }
#endif
        {   // V tile: cooperative transposed staging (TDM cannot transpose).
            const int keyl = tid >> 4;            // 0..31
            const int dblk = (tid & 15) * 8;      // 0..120
            const int jk   = jk0 + keyl;
            const int t    = (jk < WINDOW) ? ((c - 1) * WINDOW + jk)
                                           : (c * WINDOW + jk - WINDOW);
            u32x4 vv = {0u,0u,0u,0u};
#if !HAVE_TDM
            u32x4 kv = {0u,0u,0u,0u};
#endif
            if (jk >= WINDOW || c > 0) {
                const size_t base = ((size_t)b * SEQ_T + t) * D_MODEL
                                    + (size_t)h * HEAD_DIM + dblk;
                vv = *(const u32x4*)(Vb + base);
#if !HAVE_TDM
                kv = *(const u32x4*)(Kb + base);
#endif
            }
#if !HAVE_TDM
            *(u32x4*)&k_lds[keyl][dblk] = kv;
#endif
#pragma unroll
            for (int e = 0; e < 4; ++e) {
                const unsigned int w = vv[e];
                vT_lds[dblk + 2 * e][keyl]     = (ushort_t)(w & 0xffffu);
                vT_lds[dblk + 2 * e + 1][keyl] = (ushort_t)(w >> 16);
            }
        }
        __syncthreads();

        // S = Q * K^T for two 16-key tiles (K-dim = 128 in 4 WMMA steps)
        f32x8 s0 = zero, s1 = zero;
#pragma unroll
        for (int ds = 0; ds < 4; ++ds) {
            const int dbase = ds * 32 + half * 8;
            BF16Frag bk0, bk1;
            bk0.q[0] = *(const u32x4*)&k_lds[l16][dbase];
            bk0.q[1] = *(const u32x4*)&k_lds[l16][dbase + 16];
            bk1.q[0] = *(const u32x4*)&k_lds[l16 + 16][dbase];
            bk1.q[1] = *(const u32x4*)&k_lds[l16 + 16][dbase + 16];
            s0 = __builtin_amdgcn_wmma_f32_16x16x32_bf16(
                false, aq[ds].v, false, bk0.v, (short)0, s0, false, false);
            s1 = __builtin_amdgcn_wmma_f32_16x16x32_bf16(
                false, aq[ds].v, false, bk1.v, (short)0, s1, false, false);
        }

        // mask + online softmax (rows m = j + 8*half; half-wave reductions)
#pragma unroll
        for (int j = 0; j < 8; ++j) {
            const int mrow = j + half * 8;
            const int qr   = wave * 16 + mrow;
            const int key0 = step * 32 + l16;
            const int key1 = key0 + 16;
            const bool ok0 = (key0 < WINDOW) ? (c > 0) : ((key0 - WINDOW) <= qr);
            const bool ok1 = (key1 < WINDOW) ? (c > 0) : ((key1 - WINDOW) <= qr);
            float x0 = ok0 ? s0[j] * scale : -1e30f;
            float x1 = ok1 ? s1[j] * scale : -1e30f;
            float rmax = fmaxf(x0, x1);
            for (int off = 1; off < 16; off <<= 1)
                rmax = fmaxf(rmax, __shfl_xor(rmax, off, 32));
            const float mnew  = fmaxf(m_run[j], rmax);
            const float alpha = __expf(m_run[j] - mnew);
            const float p0 = __expf(x0 - mnew);
            const float p1 = __expf(x1 - mnew);
            float rs = p0 + p1;
            for (int off = 1; off < 16; off <<= 1)
                rs += __shfl_xor(rs, off, 32);
            m_run[j] = mnew;
            l_run[j] = l_run[j] * alpha + rs;
#pragma unroll
            for (int nt = 0; nt < 8; ++nt) o[nt][j] *= alpha;
            p_lds[wave][mrow][l16]      = f32_to_bf16(p0);
            p_lds[wave][mrow][l16 + 16] = f32_to_bf16(p1);
        }
        asm volatile("s_wait_dscnt 0" ::: "memory");

        // O += P * V  (P re-read in A layout from per-wave staging)
        BF16Frag pf;
        pf.q[0] = *(const u32x4*)&p_lds[wave][l16][half * 8];
        pf.q[1] = *(const u32x4*)&p_lds[wave][l16][half * 8 + 16];
#pragma unroll
        for (int nt = 0; nt < 8; ++nt) {
            BF16Frag bv;
            bv.q[0] = *(const u32x4*)&vT_lds[nt * 16 + l16][half * 8];
            bv.q[1] = *(const u32x4*)&vT_lds[nt * 16 + l16][half * 8 + 16];
            o[nt] = __builtin_amdgcn_wmma_f32_16x16x32_bf16(
                false, pf.v, false, bv.v, (short)0, o[nt], false, false);
        }
    }

    // normalize + store bf16 attention output (input to Wo projection)
#pragma unroll
    for (int j = 0; j < 8; ++j) {
        const int mrow = j + half * 8;
        const float inv = 1.0f / l_run[j];
        const size_t row = (size_t)b * SEQ_T + (size_t)c * WINDOW + wave * 16 + mrow;
#pragma unroll
        for (int nt = 0; nt < 8; ++nt) {
            Ob[row * D_MODEL + (size_t)h * HEAD_DIM + nt * 16 + l16] =
                f32_to_bf16(o[nt][j] * inv);
        }
    }
}

// ---------------------------------------------------------------------------
// Launcher
// ---------------------------------------------------------------------------
extern "C" void kernel_launch(void* const* d_in, const int* in_sizes, int n_in,
                              void* d_out, int out_size, void* d_ws, size_t ws_size,
                              hipStream_t stream) {
    (void)in_sizes; (void)n_in; (void)out_size; (void)ws_size;
    const float* x  = (const float*)d_in[0];
    const float* Wq = (const float*)d_in[1];
    const float* Wk = (const float*)d_in[2];
    const float* Wv = (const float*)d_in[3];
    const float* Wo = (const float*)d_in[4];
    float* out = (float*)d_out;

    const size_t M  = (size_t)2 * SEQ_T;          // 16384 rows
    const size_t MD = M * D_MODEL;                // 33.5M elements
    const size_t WD = (size_t)D_MODEL * D_MODEL;  // 4.2M elements

    char* ws = (char*)d_ws;
    ushort_t* xbf = (ushort_t*)ws;  ws += MD * 2;
    ushort_t* wqb = (ushort_t*)ws;  ws += WD * 2;
    ushort_t* wkb = (ushort_t*)ws;  ws += WD * 2;
    ushort_t* wvb = (ushort_t*)ws;  ws += WD * 2;
    ushort_t* wob = (ushort_t*)ws;  ws += WD * 2;
    ushort_t* qb  = (ushort_t*)ws;  ws += MD * 2;
    ushort_t* kb  = (ushort_t*)ws;  ws += MD * 2;
    ushort_t* vb  = (ushort_t*)ws;  ws += MD * 2;
    ushort_t* aob = (ushort_t*)ws;  ws += MD * 2;

    cvt_f32_to_bf16<<<4096, 256, 0, stream>>>(x,  xbf, (int)MD);
    cvt_f32_to_bf16<<<1024, 256, 0, stream>>>(Wq, wqb, (int)WD);
    cvt_f32_to_bf16<<<1024, 256, 0, stream>>>(Wk, wkb, (int)WD);
    cvt_f32_to_bf16<<<1024, 256, 0, stream>>>(Wv, wvb, (int)WD);
    cvt_f32_to_bf16<<<1024, 256, 0, stream>>>(Wo, wob, (int)WD);

    const dim3 gg(D_MODEL / 64, (unsigned)(M / 64));   // 32 x 256
    gemm_bf16_bt<<<gg, 128, 0, stream>>>(xbf, wqb, qb, nullptr, D_MODEL, 1);
    gemm_bf16_bt<<<gg, 128, 0, stream>>>(xbf, wkb, kb, nullptr, D_MODEL, 1);
    gemm_bf16_bt<<<gg, 128, 0, stream>>>(xbf, wvb, vb, nullptr, D_MODEL, 0);

    attn_swa<<<dim3(N_HEAD, SEQ_T / WINDOW, 2), 512, 0, stream>>>(qb, kb, vb, aob);

    gemm_bf16_bt<<<gg, 128, 0, stream>>>(aob, wob, nullptr, out, D_MODEL, 0);
}
